// MaskedMultiHeadSelfAttention_40535901339768
// MI455X (gfx1250) — compile-verified, matrix operand order checked by reading, not run on hardware
//
#include <hip/hip_runtime.h>

#define S_LEN  4096
#define DM     1024
#define NHEAD  16
#define DH     64

typedef __attribute__((ext_vector_type(16))) __bf16        v16bf;
typedef __attribute__((ext_vector_type(8)))  float         v8f;
typedef __attribute__((ext_vector_type(8)))  unsigned int  v8u;

__device__ __forceinline__ unsigned short f2bf(float f) {
  unsigned int u = __float_as_uint(f);
  u += 0x7FFFu + ((u >> 16) & 1u);          // round-to-nearest-even
  return (unsigned short)(u >> 16);
}

// Load a 16x32 bf16 fragment in the CDNA5 16-bit A-matrix layout from a
// row-major [rows][ld] bf16 (ushort) matrix, tile origin (row0, k0).
// Lanes 0-15: rows row0..row0+15, K = {0..7, 16..23};
// lanes 16-31: same rows,          K = {8..15, 24..31}.  Pairs of K are
// contiguous, so each VGPR is one aligned 32-bit load (merged to b128 by LLVM).
__device__ __forceinline__ v16bf load_frag(const unsigned short* __restrict__ base,
                                           int ld, int row0, int k0) {
  int lane = threadIdx.x & 31;
  int half = lane >> 4;
  const unsigned short* p = base + (row0 + (lane & 15)) * ld + k0 + half * 8;
  v8u u;
#pragma unroll
  for (int v = 0; v < 4; ++v) u[v]     = *(const unsigned int*)(p + 2 * v);
#pragma unroll
  for (int v = 0; v < 4; ++v) u[4 + v] = *(const unsigned int*)(p + 16 + 2 * v);
  return __builtin_bit_cast(v16bf, u);
}

__device__ __forceinline__ float redmax16(float x) {
#pragma unroll
  for (int m = 1; m < 16; m <<= 1) x = fmaxf(x, __shfl_xor(x, m, 32));
  return x;
}
__device__ __forceinline__ float redsum16(float x) {
#pragma unroll
  for (int m = 1; m < 16; m <<= 1) x += __shfl_xor(x, m, 32);
  return x;
}

// ---------------- conversion kernels ----------------
__global__ void cvt_copy(const float* __restrict__ src, unsigned short* __restrict__ dst, int n) {
  int i = blockIdx.x * blockDim.x + threadIdx.x;
  if (i < n) dst[i] = f2bf(__builtin_nontemporal_load(src + i));
}
// src is [K][N] row-major fp32 -> dst is [N][K] row-major bf16
__global__ void cvt_T(const float* __restrict__ src, unsigned short* __restrict__ dst, int K, int N) {
  int i = blockIdx.x * blockDim.x + threadIdx.x;
  if (i < K * N) {
    int k = i / N, n = i % N;
    dst[n * K + k] = f2bf(__builtin_nontemporal_load(src + i));
  }
}

// ---------------- QKV projection ----------------
// Each wave computes a 16x64 strip of qkv = x @ w_qkv + b, then scatters
// Q,K -> [H][S][64] and V -> transposed [H][64][S] in bf16.
__global__ __launch_bounds__(256) void qkv_gemm(
    const unsigned short* __restrict__ xb,      // [S][DM] bf16
    const unsigned short* __restrict__ wT,      // [3*DM][DM] bf16 (transposed)
    const float* __restrict__ bias,             // [3*DM]
    unsigned short* __restrict__ qb,            // [H][S][DH]
    unsigned short* __restrict__ kb,            // [H][S][DH]
    unsigned short* __restrict__ vT)            // [H][DH][S]
{
  int wave = (blockIdx.x * blockDim.x + threadIdx.x) >> 5;
  int lane = threadIdx.x & 31;
  int mt = wave % (S_LEN / 16);
  int ng = wave / (S_LEN / 16);
  int m0 = mt * 16, n0 = ng * 64;

  v8f acc[4] = {};
  for (int k0 = 0; k0 < DM; k0 += 32) {
    v16bf a = load_frag(xb, DM, m0, k0);
#pragma unroll
    for (int t = 0; t < 4; ++t) {
      v16bf b = load_frag(wT, DM, n0 + t * 16, k0);
      acc[t] = __builtin_amdgcn_wmma_f32_16x16x32_bf16(false, a, false, b,
                                                       (short)0, acc[t], false, false);
    }
  }

  int half = lane >> 4;
  int sec  = n0 / DM;                 // 0=Q 1=K 2=V (n0 is 64-aligned: whole strip same section/head)
  int h    = (n0 % DM) / DH;
#pragma unroll
  for (int t = 0; t < 4; ++t) {
    int n = n0 + t * 16 + (lane & 15);
    int d = n % DH;
    float bv = bias[n];
#pragma unroll
    for (int i = 0; i < 8; ++i) {
      int m = m0 + i + half * 8;
      unsigned short val = f2bf(acc[t][i] + bv);
      if (sec == 0)      qb[(h * S_LEN + m) * DH + d] = val;
      else if (sec == 1) kb[(h * S_LEN + m) * DH + d] = val;
      else               vT[(h * DH + d) * S_LEN + m] = val;
    }
  }
}

// ---------------- causal flash attention ----------------
// One 32-key block: QK^T (2 WMMA/tile), online softmax, P via LDS -> A-frag,
// P.V (4 WMMA).  MASKED=false skips the causal compare entirely (blocks fully
// below the diagonal) -- that is ~97% of all blocks.
template <bool MASKED>
__device__ __forceinline__ void attn_block(
    int t0, int m0, int half, int col, int m0p15,
    const unsigned short* __restrict__ kh,
    const unsigned short* __restrict__ vh,
    v16bf qf0, v16bf qf1,
    v8f (&accO)[4], float (&mrow)[8], float (&lrow)[8],
    unsigned short* ldsPw)
{
  int lane = threadIdx.x & 31;
  // prefetch next block's K rows and V rows into near cache (L2-resident data)
  if (t0 + 32 <= m0p15) {
    __builtin_prefetch(kh + (t0 + 32 + lane) * DH, 0, 3);
    __builtin_prefetch(vh + lane * S_LEN + t0 + 32, 0, 3);
    __builtin_prefetch(vh + (lane + 32) * S_LEN + t0 + 32, 0, 3);
  }

  v8f s[2] = {};
#pragma unroll
  for (int tt = 0; tt < 2; ++tt) {
    v16bf b0 = load_frag(kh, DH, t0 + tt * 16, 0);
    s[tt] = __builtin_amdgcn_wmma_f32_16x16x32_bf16(false, qf0, false, b0,
                                                    (short)0, s[tt], false, false);
    v16bf b1 = load_frag(kh, DH, t0 + tt * 16, 32);
    s[tt] = __builtin_amdgcn_wmma_f32_16x16x32_bf16(false, qf1, false, b1,
                                                    (short)0, s[tt], false, false);
  }
  // scale (1/sqrt(64)) + causal mask (diagonal blocks only)
#pragma unroll
  for (int tt = 0; tt < 2; ++tt) {
    int tc = t0 + tt * 16 + col;
#pragma unroll
    for (int i = 0; i < 8; ++i) {
      float v = s[tt][i] * 0.125f;
      if (MASKED) {
        int row = m0 + i + half * 8;
        v = (tc > row) ? -__builtin_inff() : v;
      }
      s[tt][i] = v;
    }
  }
  float alpha[8];
#pragma unroll
  for (int i = 0; i < 8; ++i) {
    float mx   = redmax16(fmaxf(s[0][i], s[1][i]));
    float mnew = fmaxf(mrow[i], mx);
    alpha[i] = __expf(mrow[i] - mnew);       // 0 on first block (exp(-inf))
    mrow[i]  = mnew;
    float p0 = __expf(s[0][i] - mnew);
    float p1 = __expf(s[1][i] - mnew);
    lrow[i] = lrow[i] * alpha[i] + redsum16(p0 + p1);
    ldsPw[(i + half * 8) * 32 + col]      = f2bf(p0);
    ldsPw[(i + half * 8) * 32 + 16 + col] = f2bf(p1);
  }
  asm volatile("s_wait_dscnt 0x0" ::: "memory");
  v16bf pf = load_frag(ldsPw, 32, 0, 0);
#pragma unroll
  for (int t = 0; t < 4; ++t) {
#pragma unroll
    for (int i = 0; i < 8; ++i) accO[t][i] *= alpha[i];
    v16bf vf = load_frag(vh, S_LEN, t * 16, t0);
    accO[t] = __builtin_amdgcn_wmma_f32_16x16x32_bf16(false, pf, false, vf,
                                                      (short)0, accO[t], false, false);
  }
}

__global__ __launch_bounds__(256) void flash_attn(
    const unsigned short* __restrict__ qb,   // [H][S][DH]
    const unsigned short* __restrict__ kb,   // [H][S][DH]
    const unsigned short* __restrict__ vT,   // [H][DH][S]
    unsigned short* __restrict__ ao)         // [S][DM] bf16
{
  __shared__ unsigned short ldsP[8][16 * 32];
  int wv   = threadIdx.x >> 5;
  int lane = threadIdx.x & 31;
  int wave = blockIdx.x * 8 + wv;
  int h  = wave / (S_LEN / 16);
  int qt = wave % (S_LEN / 16);
  int m0 = qt * 16;
  int half = lane >> 4;
  int col  = lane & 15;
  unsigned short* ldsPw = &ldsP[wv][0];

  const unsigned short* qh = qb + h * S_LEN * DH;
  const unsigned short* kh = kb + h * S_LEN * DH;
  const unsigned short* vh = vT + h * DH * S_LEN;

  v16bf qf0 = load_frag(qh, DH, m0, 0);
  v16bf qf1 = load_frag(qh, DH, m0, 32);

  v8f accO[4] = {};
  float mrow[8], lrow[8];
#pragma unroll
  for (int i = 0; i < 8; ++i) { mrow[i] = -__builtin_inff(); lrow[i] = 0.f; }

  int jum  = m0 / 32;                 // blocks [0, jum): fully below diagonal
  int jmax = (m0 + 15) / 32;          // causal last block
  for (int j = 0; j < jum; ++j)
    attn_block<false>(j * 32, m0, half, col, m0 + 15, kh, vh, qf0, qf1,
                      accO, mrow, lrow, ldsPw);
  for (int j = jum; j <= jmax; ++j)
    attn_block<true>(j * 32, m0, half, col, m0 + 15, kh, vh, qf0, qf1,
                     accO, mrow, lrow, ldsPw);

  // normalize and store bf16 [S][DM]
#pragma unroll
  for (int t = 0; t < 4; ++t) {
#pragma unroll
    for (int i = 0; i < 8; ++i) {
      int m = m0 + i + half * 8;
      ao[m * DM + h * DH + t * 16 + col] = f2bf(accO[t][i] / lrow[i]);
    }
  }
}

// ---------------- output projection ----------------
__global__ __launch_bounds__(256) void out_gemm(
    const unsigned short* __restrict__ ab,   // [S][DM] bf16
    const unsigned short* __restrict__ wT,   // [DM][DM] bf16 (transposed)
    const float* __restrict__ bias,          // [DM]
    float* __restrict__ out)                 // [S][DM] fp32
{
  int wave = (blockIdx.x * blockDim.x + threadIdx.x) >> 5;
  int lane = threadIdx.x & 31;
  int mt = wave % (S_LEN / 16);
  int ng = wave / (S_LEN / 16);
  int m0 = mt * 16, n0 = ng * 64;

  v8f acc[4] = {};
  for (int k0 = 0; k0 < DM; k0 += 32) {
    v16bf a = load_frag(ab, DM, m0, k0);
#pragma unroll
    for (int t = 0; t < 4; ++t) {
      v16bf b = load_frag(wT, DM, n0 + t * 16, k0);
      acc[t] = __builtin_amdgcn_wmma_f32_16x16x32_bf16(false, a, false, b,
                                                       (short)0, acc[t], false, false);
    }
  }
  int half = lane >> 4;
#pragma unroll
  for (int t = 0; t < 4; ++t) {
    int n = n0 + t * 16 + (lane & 15);
    float bv = bias[n];
#pragma unroll
    for (int i = 0; i < 8; ++i)   // output written once, never reread: bypass-persist
      __builtin_nontemporal_store(acc[t][i] + bv, &out[(m0 + i + half * 8) * DM + n]);
  }
}

extern "C" void kernel_launch(void* const* d_in, const int* in_sizes, int n_in,
                              void* d_out, int out_size, void* d_ws, size_t ws_size,
                              hipStream_t stream) {
  const float* x     = (const float*)d_in[0];   // [S, DM]
  const float* w_qkv = (const float*)d_in[1];   // [DM, 3DM]
  const float* b_qkv = (const float*)d_in[2];   // [3DM]
  const float* w_o   = (const float*)d_in[3];   // [DM, DM]
  const float* b_o   = (const float*)d_in[4];   // [DM]

  char* ws = (char*)d_ws;
  unsigned short* xb    = (unsigned short*)(ws);                        // 8 MB
  unsigned short* wqkvT = (unsigned short*)(ws + ( 8ull << 20));        // 6 MB
  unsigned short* woT   = (unsigned short*)(ws + (14ull << 20));        // 2 MB
  unsigned short* qb    = (unsigned short*)(ws + (16ull << 20));        // 8 MB
  unsigned short* kb    = (unsigned short*)(ws + (24ull << 20));        // 8 MB
  unsigned short* vT    = (unsigned short*)(ws + (32ull << 20));        // 8 MB
  unsigned short* ao    = xb;   // alias: x-bf16 is dead after qkv_gemm

  cvt_copy<<<(S_LEN * DM + 255) / 256, 256, 0, stream>>>(x, xb, S_LEN * DM);
  cvt_T   <<<(DM * 3 * DM + 255) / 256, 256, 0, stream>>>(w_qkv, wqkvT, DM, 3 * DM);
  cvt_T   <<<(DM * DM + 255) / 256, 256, 0, stream>>>(w_o, woT, DM, DM);

  // 256 M-tiles * 48 N-strips = 12288 waves -> 1536 blocks of 8 waves
  qkv_gemm<<<(256 * 48) / 8, 256, 0, stream>>>(xb, wqkvT, b_qkv, qb, kb, vT);
  // 16 heads * 256 q-tiles = 4096 waves -> 512 blocks
  flash_attn<<<(NHEAD * 256) / 8, 256, 0, stream>>>(qb, kb, vT, ao);
  // 256 M-tiles * 16 N-strips = 4096 waves -> 512 blocks
  out_gemm<<<(256 * 16) / 8, 256, 0, stream>>>(ao, woT, b_o, (float*)d_out);
}
